// EfficientMemoryHadamard_30726196035787
// MI455X (gfx1250) — matrix operand order repaired and checked by reading, not compiled, not run on hardware
//
#include <hip/hip_runtime.h>

#define B_    8
#define D_    2048
#define PLANE (D_ * D_)          // 4194304
#define NTOT  (B_ * PLANE)       // 33554432
#define RK    16
#define NBINS 4096

typedef __attribute__((ext_vector_type(16))) __bf16 v16bf;
typedef __attribute__((ext_vector_type(8)))  float  v8f;

union BfFrag {
    v16bf          v;
    unsigned short u[16];
    unsigned int   w[8];
};

static __device__ __forceinline__ unsigned short f2bf(float f) {
    unsigned u = __float_as_uint(f);
    unsigned r = u + 0x7FFFu + ((u >> 16) & 1u);   // round-to-nearest-even
    return (unsigned short)(r >> 16);
}

// ---------------------------------------------------------------- Hadamard
__global__ void mul_kernel(const float4* __restrict__ a,
                           const float4* __restrict__ b,
                           float4* __restrict__ o) {
    size_t i = (size_t)blockIdx.x * blockDim.x + threadIdx.x;
    __builtin_prefetch(&a[i + 4096], 0, 0);
    __builtin_prefetch(&b[i + 4096], 0, 0);
    float4 va = a[i], vb = b[i];
    float4 r;
    r.x = va.x * vb.x; r.y = va.y * vb.y; r.z = va.z * vb.z; r.w = va.w * vb.w;
    o[i] = r;
}

// ---------------------------------------------------------------- init ws
__global__ void init_kernel(unsigned* hist, unsigned* scal) {
    int i = blockIdx.x * blockDim.x + threadIdx.x;
    if (i < NBINS) hist[i] = 0u;
    if (i < 4)     scal[i] = 0u;
}

// ---------------------------------------------------------------- abs-max of x[0]
__global__ void absmax_kernel(const float4* __restrict__ x, unsigned* amax) {
    __shared__ float red[256];
    int t = threadIdx.x;
    int i = blockIdx.x * 256 + t;
    float4 v = x[i];
    float m = fmaxf(fmaxf(fabsf(v.x), fabsf(v.y)), fmaxf(fabsf(v.z), fabsf(v.w)));
    red[t] = m; __syncthreads();
    for (int s = 128; s > 0; s >>= 1) {
        if (t < s) red[t] = fmaxf(red[t], red[t + s]);
        __syncthreads();
    }
    if (t == 0) atomicMax(amax, __float_as_uint(red[0]));
}

// ---------------------------------------------------------------- histogram of |x[0]|
__global__ void hist_kernel(const float4* __restrict__ x,
                            const unsigned* __restrict__ amaxp,
                            unsigned* __restrict__ hist) {
    __shared__ unsigned h[NBINS];
    int t = threadIdx.x;
    for (int i = t; i < NBINS; i += 256) h[i] = 0u;
    __syncthreads();
    float inv = (float)NBINS / fmaxf(__uint_as_float(*amaxp), 1e-30f);
    int i = blockIdx.x * 256 + t;
    float4 v = x[i];
    float a0 = fabsf(v.x), a1 = fabsf(v.y), a2 = fabsf(v.z), a3 = fabsf(v.w);
    int b0 = min((int)(a0 * inv), NBINS - 1);
    int b1 = min((int)(a1 * inv), NBINS - 1);
    int b2 = min((int)(a2 * inv), NBINS - 1);
    int b3 = min((int)(a3 * inv), NBINS - 1);
    atomicAdd(&h[b0], 1u); atomicAdd(&h[b1], 1u);
    atomicAdd(&h[b2], 1u); atomicAdd(&h[b3], 1u);
    __syncthreads();
    for (int i2 = t; i2 < NBINS; i2 += 256)
        if (h[i2]) atomicAdd(&hist[i2], h[i2]);
}

// ---------------------------------------------------------------- quantile scan
__global__ void quantile_kernel(const unsigned* __restrict__ hist,
                                const unsigned* __restrict__ amaxp,
                                float* ws_thresh, float* out_o) {
    float amax = __uint_as_float(*amaxp);
    unsigned target = (unsigned)(0.99f * (float)PLANE);
    unsigned cum = 0; int bin = NBINS - 1;
    for (int i = 0; i < NBINS; ++i) {
        cum += hist[i];
        if (cum >= target) { bin = i; break; }
    }
    float th = amax * (float)(bin + 1) * (1.0f / (float)NBINS);
    *ws_thresh = th;
    *out_o = th;
}

// -------------------------------------------------- thresholded mean + bf16 (A and A^T copies)
__global__ void mean_bf_kernel(const float* __restrict__ x,
                               const float* __restrict__ threshp,
                               float* __restrict__ A,
                               unsigned short* __restrict__ Abf,
                               unsigned short* __restrict__ Atbf) {
    size_t i4 = ((size_t)blockIdx.x * 256 + threadIdx.x) * 4;
    float th = *threshp;
    float s0 = 0.f, s1 = 0.f, s2 = 0.f, s3 = 0.f;
    for (int b = 0; b < B_; ++b) {
        float4 v = *(const float4*)(x + (size_t)b * PLANE + i4);
        s0 += (fabsf(v.x) > th) ? 0.f : v.x;
        s1 += (fabsf(v.y) > th) ? 0.f : v.y;
        s2 += (fabsf(v.z) > th) ? 0.f : v.z;
        s3 += (fabsf(v.w) > th) ? 0.f : v.w;
    }
    s0 *= 0.125f; s1 *= 0.125f; s2 *= 0.125f; s3 *= 0.125f;
    A[i4 + 0] = s0; A[i4 + 1] = s1; A[i4 + 2] = s2; A[i4 + 3] = s3;
    unsigned short b0 = f2bf(s0), b1 = f2bf(s1), b2 = f2bf(s2), b3 = f2bf(s3);
    Abf[i4 + 0] = b0; Abf[i4 + 1] = b1; Abf[i4 + 2] = b2; Abf[i4 + 3] = b3;
    int i = (int)(i4 >> 11), j = (int)(i4 & (D_ - 1));
    Atbf[(size_t)(j + 0) * D_ + i] = b0;
    Atbf[(size_t)(j + 1) * D_ + i] = b1;
    Atbf[(size_t)(j + 2) * D_ + i] = b2;
    Atbf[(size_t)(j + 3) * D_ + i] = b3;
}

// ---------------------------------------------------------------- deterministic random panel
__global__ void randinit_kernel(float* __restrict__ V) {
    int i = blockIdx.x * 256 + threadIdx.x;            // 2048*16
    unsigned h = (unsigned)i * 2654435761u + 0x9E3779B9u;
    h ^= h >> 16; h *= 2246822519u; h ^= h >> 13; h *= 3266489917u; h ^= h >> 16;
    V[i] = (float)(h & 0xFFFFFFu) * (1.0f / 8388608.0f) - 1.0f;
}

// ---------------------------------------------------------------- modified Gram-Schmidt (2048x16)
__global__ void orth_kernel(float* __restrict__ M) {
    __shared__ float red[256];
    __shared__ float sval;
    int t = threadIdx.x;
    for (int j = 0; j < RK; ++j) {
        float p = 0.f;
        for (int r = t; r < D_; r += 256) { float v = M[r * RK + j]; p += v * v; }
        red[t] = p; __syncthreads();
        for (int s = 128; s > 0; s >>= 1) { if (t < s) red[t] += red[t + s]; __syncthreads(); }
        if (t == 0) sval = rsqrtf(fmaxf(red[0], 1e-20f));
        __syncthreads();
        float inv = sval;
        for (int r = t; r < D_; r += 256) M[r * RK + j] *= inv;
        __syncthreads();
        for (int k = j + 1; k < RK; ++k) {
            float d = 0.f;
            for (int r = t; r < D_; r += 256) d += M[r * RK + j] * M[r * RK + k];
            red[t] = d; __syncthreads();
            for (int s = 128; s > 0; s >>= 1) { if (t < s) red[t] += red[t + s]; __syncthreads(); }
            if (t == 0) sval = red[0];
            __syncthreads();
            float dj = sval;
            for (int r = t; r < D_; r += 256) M[r * RK + k] -= dj * M[r * RK + j];
            __syncthreads();
        }
    }
}

// ------------------------------------------- panel f32 (2048x16) -> transposed bf16 (16x2048)
__global__ void tobf_t_kernel(const float* __restrict__ src, unsigned short* __restrict__ dst) {
    int i = blockIdx.x * 256 + threadIdx.x;   // over 16*2048, output-coalesced
    int j = i >> 11, r = i & (D_ - 1);
    dst[i] = f2bf(src[r * RK + j]);
}

// ---------------------------------------------------------------- tall-skinny GEMM via WMMA
// Y(2048x16,f32) = A(2048x2048,bf16 row-major) * X  with X given transposed (16x2048 bf16).
// One wave per 16x16 output tile; all loads are aligned b32 of contiguous bf16 pairs.
__global__ void gemm_wmma_kernel(const unsigned short* __restrict__ Abf,
                                 const unsigned short* __restrict__ Xtbf,
                                 float* __restrict__ Y) {
    int tile   = blockIdx.x;          // 0..127
    int lane   = threadIdx.x;         // 0..31, full wave active (EXEC all-1s)
    int mlocal = lane & 15;
    int half   = lane >> 4;
    int m      = tile * 16 + mlocal;
    const unsigned int* A32 = (const unsigned int*)Abf;
    const unsigned int* X32 = (const unsigned int*)Xtbf;

    v8f c = {};
    for (int k0 = 0; k0 < D_; k0 += 32) {
        BfFrag a, b;
#pragma unroll
        for (int v = 0; v < 8; ++v) {
            // A fragment (16x32 bf16): lane row = mlocal, K block = half*16
            int kbA = k0 + half * 16 + 2 * v;
            a.w[v] = A32[(m * D_ + kbA) >> 1];
            // B fragment (32x16 bf16): lane col = mlocal, K block = half*16
            int kbB = k0 + half * 16 + 2 * v;
            b.w[v] = X32[(mlocal * D_ + kbB) >> 1];
        }
        c = __builtin_amdgcn_wmma_f32_16x16x32_bf16(false, a.v, false, b.v,
                                                    (short)0, c, false, false);
    }
    // D layout: VGPR r -> row = r + 8*half, col = mlocal
    int rb = tile * 16 + 8 * half;
#pragma unroll
    for (int r = 0; r < 8; ++r)
        Y[(rb + r) * RK + mlocal] = c[r];
}

// ---------------------------------------------------------------- R = V^T
__global__ void copyR_kernel(const float* __restrict__ V, float* __restrict__ R) {
    int i = blockIdx.x * 256 + threadIdx.x;   // 16*2048
    int j = i >> 11, cidx = i & (D_ - 1);
    R[i] = V[cidx * RK + j];
}

// ---------------------------------------------------------------- max |x_res - L R|
__global__ void maxdiff_kernel(const float* __restrict__ x,
                               const float* __restrict__ L,
                               const float* __restrict__ R,
                               const float* __restrict__ threshp,
                               unsigned* __restrict__ maxslot) {
    __shared__ float red[256];
    size_t idx = (size_t)blockIdx.x * 256 + threadIdx.x;   // over PLANE
    int i = (int)(idx >> 11), j = (int)(idx & (D_ - 1));
    float th = *threshp;
    float lr = 0.f;
#pragma unroll
    for (int k = 0; k < RK; ++k) lr += L[i * RK + k] * R[k * D_ + j];
    float m = 0.f;
    for (int b = 0; b < B_; ++b) {
        float v  = x[(size_t)b * PLANE + idx];
        float xr = (fabsf(v) > th) ? 0.f : v;
        m = fmaxf(m, fabsf(xr - lr));
    }
    int t = threadIdx.x;
    red[t] = m; __syncthreads();
    for (int s = 128; s > 0; s >>= 1) { if (t < s) red[t] = fmaxf(red[t], red[t + s]); __syncthreads(); }
    if (t == 0) atomicMax(maxslot, __float_as_uint(red[0]));
}

// ---------------------------------------------------------------- scalars
__global__ void finalize_kernel(const unsigned* __restrict__ maxu, float* __restrict__ sc) {
    *sc = __uint_as_float(*maxu) * (1.0f / 127.0f);
}

// ================================================================ host side
extern "C" void kernel_launch(void* const* d_in, const int* in_sizes, int n_in,
                              void* d_out, int out_size, void* d_ws, size_t ws_size,
                              hipStream_t stream) {
    const float* xs[2] = { (const float*)d_in[0], (const float*)d_in[1] };
    float* out = (float*)d_out;
    char*  ws  = (char*)d_ws;

    // workspace layout (~34 MB)
    float*          A    = (float*)(ws + 0);                     // 2048x2048 f32
    unsigned short* Abf  = (unsigned short*)(ws + 16777216);     // 2048x2048 bf16 row-major
    unsigned short* Atbf = (unsigned short*)(ws + 25165824);     // 2048x2048 bf16 transposed
    float*          V    = (float*)(ws + 33554432);              // 2048x16 f32
    float*          U    = (float*)(ws + 33685504);              // 2048x16 f32
    unsigned short* Vtbf = (unsigned short*)(ws + 33816576);     // 16x2048 bf16
    unsigned short* Utbf = (unsigned short*)(ws + 33882112);     // 16x2048 bf16
    unsigned*       hist = (unsigned*)(ws + 33947648);
    unsigned*       scal = (unsigned*)(ws + 33964032);           // [0]=amax [1]=maxdiff [2]=thresh
    float*          thr  = (float*)&scal[2];

    // output offsets (flat tuple concat: result,o1,L1,R1,sc1,o2,L2,R2,sc2)
    const size_t O_[2]  = { 33554432u, 33619970u };
    const size_t L_[2]  = { 33554433u, 33619971u };
    const size_t R_[2]  = { 33587201u, 33652739u };
    const size_t SC_[2] = { 33619969u, 33685507u };

    // result = x1 * x2  (pure bandwidth)
    mul_kernel<<<NTOT / 4 / 256, 256, 0, stream>>>(
        (const float4*)xs[0], (const float4*)xs[1], (float4*)out);

    for (int t = 0; t < 2; ++t) {
        const float* x = xs[t];

        init_kernel<<<NBINS / 256, 256, 0, stream>>>(hist, scal);
        absmax_kernel<<<PLANE / 4 / 256, 256, 0, stream>>>((const float4*)x, &scal[0]);
        hist_kernel<<<PLANE / 4 / 256, 256, 0, stream>>>((const float4*)x, &scal[0], hist);
        quantile_kernel<<<1, 1, 0, stream>>>(hist, &scal[0], thr, out + O_[t]);

        mean_bf_kernel<<<PLANE / 4 / 256, 256, 0, stream>>>(x, thr, A, Abf, Atbf);

        // randomized subspace iteration for rank-16 factors (L2-resident WMMA GEMMs)
        randinit_kernel<<<D_ * RK / 256, 256, 0, stream>>>(V);
        orth_kernel<<<1, 256, 0, stream>>>(V);
        tobf_t_kernel<<<D_ * RK / 256, 256, 0, stream>>>(V, Vtbf);
        for (int it = 0; it < 2; ++it) {
            gemm_wmma_kernel<<<D_ / 16, 32, 0, stream>>>(Abf, Vtbf, U);    // U = A V
            orth_kernel<<<1, 256, 0, stream>>>(U);
            tobf_t_kernel<<<D_ * RK / 256, 256, 0, stream>>>(U, Utbf);
            gemm_wmma_kernel<<<D_ / 16, 32, 0, stream>>>(Atbf, Utbf, V);   // V = A^T U
            orth_kernel<<<1, 256, 0, stream>>>(V);
            tobf_t_kernel<<<D_ * RK / 256, 256, 0, stream>>>(V, Vtbf);
        }
        gemm_wmma_kernel<<<D_ / 16, 32, 0, stream>>>(Abf, Vtbf, out + L_[t]); // L = A V
        copyR_kernel<<<RK * D_ / 256, 256, 0, stream>>>(V, out + R_[t]);      // R = V^T

        maxdiff_kernel<<<PLANE / 256, 256, 0, stream>>>(x, out + L_[t], out + R_[t],
                                                        thr, &scal[1]);
        finalize_kernel<<<1, 1, 0, stream>>>(&scal[1], out + SC_[t]);
    }
    (void)in_sizes; (void)n_in; (void)out_size; (void)ws_size;
}
